// DAWN_34213709480502
// MI455X (gfx1250) — compile-verified, matrix operand order checked by reading, not run on hardware
//
#include <hip/hip_runtime.h>
#include <hip/hip_bf16.h>

// ---------------- problem constants (from reference setup_inputs) -----------
constexpr int BB   = 4;
constexpr int S    = 1024;
constexpr int D    = 768;
constexpr int T    = BB * S;      // 4096 tokens
constexpr int DSP  = 128;         // d_space
constexpr int NQK  = 1024;
constexpr int NV   = 1024;
constexpr int NKN  = 4096;
constexpr int NEMB = NQK + NV + NKN; // 6144
constexpr int H    = 12;
constexpr int DH   = 64;          // D / H
constexpr int KQK  = 32;          // MAX_K_QK / MAX_K_V
constexpr int KKN  = 64;          // MAX_K_KNOW

typedef __bf16 bf16_t;
typedef __attribute__((ext_vector_type(16))) __bf16 v16bf;
typedef __attribute__((ext_vector_type(8)))  float  v8f;
typedef __attribute__((ext_vector_type(4)))  int    v4i;
typedef __attribute__((address_space(1))) v4i* gv4i_p;   // global (device) AS
typedef __attribute__((address_space(3))) v4i* lv4i_p;   // LDS AS

// ---------------- small helpers --------------------------------------------
__device__ inline float block_reduce_sum(float v, float* red) {
    int tid = threadIdx.x;
    red[tid] = v; __syncthreads();
    for (int s = blockDim.x >> 1; s > 0; s >>= 1) {
        if (tid < s) red[tid] += red[tid + s];
        __syncthreads();
    }
    float r = red[0]; __syncthreads();
    return r;
}

// ---------------- WMMA GEMM: C[M,N] = A[M,K] * Bt[N,K]^T (bf16 -> f32) ------
// blockDim = 256 (8 waves). All waves in a block share the same 16 A-rows
// (blockIdx.y); the A tile is staged in LDS in K-chunks of KC via gfx1250
// GLOBAL_LOAD_ASYNC_TO_LDS_B128 + s_wait_asynccnt. Each wave owns a 16x64
// output tile: one LDS A-fragment feeds 4 WMMAs per k-step.
constexpr int KC = 256;           // staged K-chunk: 16 x 256 bf16 = 8 KB LDS

__global__ void k_gemm_abt(const bf16_t* __restrict__ A, int lda,
                           const bf16_t* __restrict__ Bt, int ldb,
                           float* __restrict__ C, int ldc,
                           int nt64, int K,
                           const float* __restrict__ bias, float scale) {
    __shared__ bf16_t ta[16 * KC];
    int tid  = threadIdx.x;
    int wv   = tid >> 5;
    int lane = tid & 31;
    int bn   = blockIdx.x * 8 + wv;          // 64-column group owned by wave
    int bm   = blockIdx.y;
    bool active = (bn < nt64);               // wave-uniform
    int half = lane >> 4;
    int l16  = lane & 15;

    const bf16_t* b_ptr[4];
#pragma unroll
    for (int j = 0; j < 4; ++j)
        b_ptr[j] = Bt + (size_t)(bn * 64 + j * 16 + l16) * ldb + half * 16;

    v8f acc[4] = {};
    for (int kc0 = 0; kc0 < K; kc0 += KC) {
        int kcur = K - kc0 < KC ? K - kc0 : KC;
        // ---- stage A[bm*16 .. +15][kc0 .. kc0+kcur) into LDS (16B chunks) --
        int cpr    = kcur >> 3;              // 16B chunks per row
        int nchunk = 16 * cpr;
        for (int c = tid; c < nchunk; c += 256) {
            int row  = c / cpr;
            int col8 = (c - row * cpr) * 8;
            const bf16_t* g = A + (size_t)(bm * 16 + row) * lda + kc0 + col8;
            bf16_t*       l = &ta[row * KC + col8];
#if __has_builtin(__builtin_amdgcn_global_load_async_to_lds_b128)
            __builtin_amdgcn_global_load_async_to_lds_b128(
                (gv4i_p)(v4i*)(void*)const_cast<bf16_t*>(g),
                (lv4i_p)(v4i*)(void*)l, 0, 0);
#else
            *(uint4*)l = *(const uint4*)g;
#endif
        }
#if __has_builtin(__builtin_amdgcn_global_load_async_to_lds_b128)
#if __has_builtin(__builtin_amdgcn_s_wait_asynccnt)
        __builtin_amdgcn_s_wait_asynccnt(0);
#else
        asm volatile("s_wait_asynccnt 0x0" ::: "memory");
#endif
#endif
        __syncthreads();
        // ---- compute: A fragment from LDS, B fragments from global --------
        if (active) {
            for (int k0 = 0; k0 < kcur; k0 += 32) {
                v16bf a = *(const v16bf*)&ta[l16 * KC + k0 + half * 16];
                if (k0 + 32 < kcur) {
#pragma unroll
                    for (int j = 0; j < 4; ++j)      // gfx1250 global_prefetch_b8
                        __builtin_prefetch(b_ptr[j] + kc0 + k0 + 32, 0, 0);
                }
#pragma unroll
                for (int j = 0; j < 4; ++j) {
                    v16bf b = *(const v16bf*)(b_ptr[j] + kc0 + k0);
                    acc[j] = __builtin_amdgcn_wmma_f32_16x16x32_bf16(
                                 false, a, false, b, (short)0, acc[j], false, false);
                }
            }
        }
        __syncthreads();
    }
    if (active) {
#pragma unroll
        for (int j = 0; j < 4; ++j) {
            int col = bn * 64 + j * 16 + l16;
            float bv = bias ? bias[col] : 0.0f;
#pragma unroll
            for (int r = 0; r < 8; ++r) {
                int m = bm * 16 + half * 8 + r;  // D layout: half0->rows0..7
                C[(size_t)m * ldc + col] = acc[j][r] * scale + bv;
            }
        }
    }
}

// ---------------- misc elementwise / reduction kernels ----------------------
__global__ void k_zero(float* p, int n) {
    for (int i = blockIdx.x * blockDim.x + threadIdx.x; i < n;
         i += gridDim.x * blockDim.x) p[i] = 0.0f;
}

__global__ void k_add(const float* a, const float* b, float* c, int n) {
    for (int i = blockIdx.x * blockDim.x + threadIdx.x; i < n;
         i += gridDim.x * blockDim.x) c[i] = a[i] + b[i];
}

__global__ void k_cvt_bf(const float* in, bf16_t* out, int n) {
    for (int i = blockIdx.x * blockDim.x + threadIdx.x; i < n;
         i += gridDim.x * blockDim.x) out[i] = (bf16_t)in[i];
}

// transpose f32 (K,N) -> bf16 (N,K)
__global__ void k_transpose_bf(const float* in, bf16_t* out, int K, int N) {
    int i = blockIdx.x * blockDim.x + threadIdx.x;
    if (i >= K * N) return;
    int k = i / N, n = i - k * N;
    out[(size_t)n * K + k] = (bf16_t)in[i];
}

// row-normalize neuron_emb (NEMB x 128) -> bf16, blockDim = 128
__global__ void k_norm_emb(const float* emb, bf16_t* out) {
    __shared__ float red[128];
    int r = blockIdx.x, tid = threadIdx.x;
    const float* e = emb + (size_t)r * DSP;
    float v = e[tid];
    red[tid] = v * v; __syncthreads();
    for (int s = 64; s > 0; s >>= 1) {
        if (tid < s) red[tid] += red[tid + s];
        __syncthreads();
    }
    float rn = 1.0f / (sqrtf(red[0]) + 1e-8f);
    out[(size_t)r * DSP + tid] = (bf16_t)(e[tid] * rn);
}

// layer norm, one block (256) per token; emits f32 + bf16
__global__ void k_layernorm(const float* x, const float* scale, const float* bias,
                            float* xn, bf16_t* xnbf) {
    __shared__ float red[256];
    int t = blockIdx.x, tid = threadIdx.x;
    const float* xr = x + (size_t)t * D;
    float s = 0.f;
    for (int d = tid; d < D; d += 256) s += xr[d];
    float mean = block_reduce_sum(s, red) * (1.0f / D);
    float s2 = 0.f;
    for (int d = tid; d < D; d += 256) { float v = xr[d] - mean; s2 += v * v; }
    float var  = block_reduce_sum(s2, red) * (1.0f / D);
    float rstd = rsqrtf(var + 1e-6f);
    for (int d = tid; d < D; d += 256) {
        float v = (xr[d] - mean) * rstd * scale[d] + bias[d];
        xn[(size_t)t * D + d]   = v;
        xnbf[(size_t)t * D + d] = (bf16_t)v;
    }
}

// tau = xn @ W(768,ncols) + b ; one block per token
__global__ void k_tau(const float* xn, const float* w, const float* b,
                      int ncols, float* tau, int ldt) {
    __shared__ float red[256];
    int t = blockIdx.x, tid = threadIdx.x;
    for (int c = 0; c < ncols; ++c) {
        float p = 0.f;
        for (int d = tid; d < D; d += 256) p += xn[(size_t)t * D + d] * w[d * ncols + c];
        float s = block_reduce_sum(p, red);
        if (tid == 0) tau[(size_t)t * ldt + c] = s + b[c];
    }
}

// threshold_gate + exact top-k, one block (256) per token; scores row -> LDS
template <int KSEL>
__global__ void k_gate_topk(const float* __restrict__ scores,
                            const float* __restrict__ tau, int ldt, int toff,
                            int N, float* __restrict__ topv, int* __restrict__ topi,
                            float* __restrict__ acc) {
    extern __shared__ float eg[];                 // N floats
    __shared__ float rv[256]; __shared__ int ri[256];
    __shared__ float selv[KSEL]; __shared__ int seli[KSEL];
    __shared__ float bc[2];
    int t = blockIdx.x, tid = threadIdx.x;
    float tv = tau[(size_t)t * ldt + toff];
    for (int j = tid; j < N; j += 256) {
        float raw  = scores[(size_t)t * N + j] - tv;
        float gate = raw > 0.0f ? raw : 1e-8f * __expf(raw);
        eg[j] = __expf(gate) - 1.0f;
    }
    __syncthreads();
    for (int it = 0; it < KSEL; ++it) {
        float bv = -3.0e38f; int bi = N;
        for (int j = tid; j < N; j += 256) {
            float v = eg[j];
            if (v > bv) { bv = v; bi = j; }
        }
        rv[tid] = bv; ri[tid] = bi; __syncthreads();
        for (int s = 128; s > 0; s >>= 1) {
            if (tid < s) {
                if (rv[tid + s] > rv[tid] ||
                    (rv[tid + s] == rv[tid] && ri[tid + s] < ri[tid])) {
                    rv[tid] = rv[tid + s]; ri[tid] = ri[tid + s];
                }
            }
            __syncthreads();
        }
        if (tid == 0) { selv[it] = rv[0]; seli[it] = ri[0]; eg[ri[0]] = -3.0e38f; }
        __syncthreads();
    }
    if (tid == 0) {
        float sum = 0.f;
        for (int i = 0; i < KSEL; ++i) sum += selv[i];
        bc[0] = sum + 1e-8f;
        bc[1] = tanhf(selv[0]);                   // top value == row max
    }
    __syncthreads();
    if (tid < KSEL) {
        float g = selv[tid] / bc[0] * bc[1];
        topv[(size_t)t * KSEL + tid] = g;
        topi[(size_t)t * KSEL + tid] = seli[tid];
        atomicAdd(&acc[seli[tid]], g);            // for aux-loss means
    }
}

// sparse sense/emit: out = sum_k (xn . emb[idx_k]) * gate_k * w[idx_k]
template <int KSEL>
__global__ void k_sense_emit(const float* __restrict__ xn,
                             const float* __restrict__ topv,
                             const int* __restrict__ topi,
                             const float* __restrict__ emb,
                             const float* __restrict__ w,
                             float* __restrict__ out) {
    __shared__ float red[256];
    __shared__ float act[KSEL];
    __shared__ int   idx[KSEL];
    int t = blockIdx.x, tid = threadIdx.x;
    if (tid < KSEL) idx[tid] = topi[(size_t)t * KSEL + tid];
    __syncthreads();
    const float* xr = xn + (size_t)t * D;
    for (int kk = 0; kk < KSEL; ++kk) {
        const float* er = emb + (size_t)idx[kk] * D;
        float p = 0.f;
        for (int d = tid; d < D; d += 256) p += xr[d] * er[d];
        float s = block_reduce_sum(p, red);
        if (tid == 0) act[kk] = s * topv[(size_t)t * KSEL + kk];
    }
    __syncthreads();
    for (int d = tid; d < D; d += 256) {
        float o = 0.f;
        for (int kk = 0; kk < KSEL; ++kk) o += act[kk] * w[(size_t)idx[kk] * D + d];
        out[(size_t)t * D + d] = o;
    }
}

// pack Q,K (B,S,D) f32 -> (B,H,S,64) bf16 ; V -> transposed (B,H,64,S) bf16
__global__ void k_pack_qkv(const float* Q, const float* K, const float* V,
                           bf16_t* qbf, bf16_t* kbf, bf16_t* vtbf) {
    int i = blockIdx.x * blockDim.x + threadIdx.x;
    if (i >= T * D) return;
    int t = i / D, c = i - t * D;
    int b = t >> 10, s = t & 1023;
    int h = c >> 6,  d = c & 63;
    size_t bh = (size_t)(b * H + h);
    size_t qk = (bh * S + s) * DH + d;
    qbf[qk] = (bf16_t)Q[i];
    kbf[qk] = (bf16_t)K[i];
    vtbf[(bh * DH + d) * S + s] = (bf16_t)V[i];
}

// causal softmax over one S x S score tile (one block per row), emits bf16 P
__global__ void k_softmax_causal(const float* __restrict__ Sc, bf16_t* __restrict__ P) {
    __shared__ float red[256];
    int i = blockIdx.x, tid = threadIdx.x;
    const float* sr = Sc + (size_t)i * S;
    float m = -3.0e38f;
    for (int j = tid; j <= i; j += 256) m = fmaxf(m, sr[j]);
    red[tid] = m; __syncthreads();
    for (int s = 128; s > 0; s >>= 1) {
        if (tid < s) red[tid] = fmaxf(red[tid], red[tid + s]);
        __syncthreads();
    }
    float rm = red[0]; __syncthreads();
    float p = 0.f;
    for (int j = tid; j <= i; j += 256) p += __expf(sr[j] - rm);
    float inv = 1.0f / block_reduce_sum(p, red);
    for (int j = tid; j < S; j += 256)
        P[(size_t)i * S + j] = (bf16_t)(j <= i ? __expf(sr[j] - rm) * inv : 0.0f);
}

// aux loss: sum over neurons of (mean - 1/n)^2 * n, atomically into *out
__global__ void k_aux(const float* aQ, const float* aK, const float* aV,
                      const float* aN, float* out) {
    __shared__ float red[256];
    int i = blockIdx.x * blockDim.x + threadIdx.x;
    float c = 0.0f;
    const float invT = 1.0f / (float)T;
    if (i < NQK)              { float m = aQ[i]          * invT - 1.0f / NQK; c = m * m * NQK; }
    else if (i < 2 * NQK)     { float m = aK[i - NQK]    * invT - 1.0f / NQK; c = m * m * NQK; }
    else if (i < 3 * NQK)     { float m = aV[i - 2*NQK]  * invT - 1.0f / NV;  c = m * m * NV;  }
    else if (i < 3*NQK + NKN) { float m = aN[i - 3*NQK]  * invT - 1.0f / NKN; c = m * m * NKN; }
    float s = block_reduce_sum(c, red);
    if (threadIdx.x == 0) atomicAdd(out, s);
}

// ---------------- host driver ----------------------------------------------
extern "C" void kernel_launch(void* const* d_in, const int* in_sizes, int n_in,
                              void* d_out, int out_size, void* d_ws, size_t ws_size,
                              hipStream_t stream) {
    (void)in_sizes; (void)n_in; (void)out_size; (void)ws_size;

    const float* x          = (const float*)d_in[0];
    const float* qk_emb     = (const float*)d_in[1];
    const float* qk_w       = (const float*)d_in[2];
    const float* v_emb      = (const float*)d_in[3];
    const float* v_w        = (const float*)d_in[4];
    const float* know_emb   = (const float*)d_in[5];
    const float* know_w     = (const float*)d_in[6];
    const float* neuron_emb = (const float*)d_in[7];
    const float* proj_attn_k = (const float*)d_in[8];
    const float* proj_attn_b = (const float*)d_in[9];
    const float* proj_know_k = (const float*)d_in[10];
    const float* proj_know_b = (const float*)d_in[11];
    const float* tau_attn_k  = (const float*)d_in[12];
    const float* tau_attn_b  = (const float*)d_in[13];
    const float* tau_know_k  = (const float*)d_in[14];
    const float* tau_know_b  = (const float*)d_in[15];
    const float* expand_O    = (const float*)d_in[16];
    const float* ln1_s = (const float*)d_in[17];
    const float* ln1_b = (const float*)d_in[18];
    const float* ln2_s = (const float*)d_in[19];
    const float* ln2_b = (const float*)d_in[20];
    float* out = (float*)d_out;

    // ---- carve workspace (256B aligned) ----
    char*  base = (char*)d_ws;
    size_t off  = 0;
    auto carve = [&](size_t bytes) -> char* {
        char* p = base + off;
        off = (off + bytes + 255) & ~(size_t)255;
        return p;
    };
    bf16_t* embbf   = (bf16_t*)carve((size_t)NEMB * DSP * 2);
    float*  xn1     = (float*) carve((size_t)T * D * 4);
    bf16_t* xn1bf   = (bf16_t*)carve((size_t)T * D * 2);
    float*  hall    = (float*) carve((size_t)T * 3 * DSP * 4);
    bf16_t* hallbf  = (bf16_t*)carve((size_t)T * 3 * DSP * 2);
    float*  tau     = (float*) carve((size_t)T * 4 * 4);
    bf16_t* wattnT  = (bf16_t*)carve((size_t)3 * DSP * D * 2);
    bf16_t* wknowT  = (bf16_t*)carve((size_t)DSP * D * 2);
    bf16_t* wexpT   = (bf16_t*)carve((size_t)D * D * 2);
    float*  scores  = (float*) carve((size_t)T * NKN * 4);     // reused scratch
    bf16_t* pbf     = (bf16_t*)carve((size_t)S * S * 2);
    float*  topvQ   = (float*) carve((size_t)T * KQK * 4);
    int*    topiQ   = (int*)   carve((size_t)T * KQK * 4);
    float*  topvK   = (float*) carve((size_t)T * KQK * 4);
    int*    topiK   = (int*)   carve((size_t)T * KQK * 4);
    float*  topvV   = (float*) carve((size_t)T * KQK * 4);
    int*    topiV   = (int*)   carve((size_t)T * KQK * 4);
    float*  topvN   = (float*) carve((size_t)T * KKN * 4);
    int*    topiN   = (int*)   carve((size_t)T * KKN * 4);
    float*  Qf      = (float*) carve((size_t)T * D * 4);
    float*  Kf      = (float*) carve((size_t)T * D * 4);
    float*  Vf      = (float*) carve((size_t)T * D * 4);
    bf16_t* qbf     = (bf16_t*)carve((size_t)BB * H * S * DH * 2);
    bf16_t* kbf     = (bf16_t*)carve((size_t)BB * H * S * DH * 2);
    bf16_t* vtbf    = (bf16_t*)carve((size_t)BB * H * S * DH * 2);
    float*  acat    = (float*) carve((size_t)T * D * 4);
    bf16_t* acatbf  = (bf16_t*)carve((size_t)T * D * 2);
    float*  aout    = (float*) carve((size_t)T * D * 4);
    float*  x1      = (float*) carve((size_t)T * D * 4);
    float*  xn2     = (float*) carve((size_t)T * D * 4);
    bf16_t* xn2bf   = (bf16_t*)carve((size_t)T * D * 2);
    float*  hkn     = (float*) carve((size_t)T * DSP * 4);
    bf16_t* hknbf   = (bf16_t*)carve((size_t)T * DSP * 2);
    float*  tauk    = (float*) carve((size_t)T * 4);
    float*  kout    = (float*) carve((size_t)T * D * 4);
    float*  accQ    = (float*) carve((size_t)NQK * 4);
    float*  accK    = (float*) carve((size_t)NQK * 4);
    float*  accV    = (float*) carve((size_t)NV  * 4);
    float*  accN    = (float*) carve((size_t)NKN * 4);

    auto gemm = [&](const bf16_t* A, int lda, const bf16_t* Bt, int ldb,
                    float* C, int ldc, int M, int N, int K,
                    const float* bias, float scale) {
        int mt = M / 16, nt64 = N / 64;              // N is always a mult of 64
        dim3 g((nt64 + 7) / 8, mt);
        k_gemm_abt<<<g, 256, 0, stream>>>(A, lda, Bt, ldb, C, ldc, nt64, K, bias, scale);
    };

    // ---- 0. zero accumulators and aux output slot ----
    k_zero<<<32, 256, 0, stream>>>(accQ, 3 * NQK + NKN);   // contiguous carve
    k_zero<<<1, 32, 0, stream>>>(out + (size_t)T * D, 1);

    // ---- 1. normalize neuron embeddings, transpose weights to bf16 ----
    k_norm_emb<<<NEMB, DSP, 0, stream>>>(neuron_emb, embbf);
    k_transpose_bf<<<(D * 3 * DSP + 255) / 256, 256, 0, stream>>>(proj_attn_k, wattnT, D, 3 * DSP);
    k_transpose_bf<<<(D * DSP + 255) / 256, 256, 0, stream>>>(proj_know_k, wknowT, D, DSP);
    k_transpose_bf<<<(D * D + 255) / 256, 256, 0, stream>>>(expand_O, wexpT, D, D);

    // ---- 2. LN1, projections, tau ----
    k_layernorm<<<T, 256, 0, stream>>>(x, ln1_s, ln1_b, xn1, xn1bf);
    gemm(xn1bf, D, wattnT, D, hall, 3 * DSP, T, 3 * DSP, D, proj_attn_b, 1.0f);
    k_cvt_bf<<<(T * 3 * DSP + 255) / 256, 256, 0, stream>>>(hall, hallbf, T * 3 * DSP);
    k_tau<<<T, 256, 0, stream>>>(xn1, tau_attn_k, tau_attn_b, 3, tau, 4);

    // ---- 3. gates + sparse Q/K/V synthesis ----
    const bf16_t* lows[3]  = { embbf, embbf, embbf + (size_t)NQK * DSP };
    const float*  embs[3]  = { qk_emb, qk_emb, v_emb };
    const float*  ws_[3]   = { qk_w, qk_w, v_w };
    float* tvs[3] = { topvQ, topvK, topvV };
    int*   tis[3] = { topiQ, topiK, topiV };
    float* accs[3] = { accQ, accK, accV };
    float* dsts[3] = { Qf, Kf, Vf };
    for (int sec = 0; sec < 3; ++sec) {
        gemm(hallbf + sec * DSP, 3 * DSP, lows[sec], DSP, scores, NQK,
             T, NQK, DSP, nullptr, 1.0f);
        k_gate_topk<KQK><<<T, 256, NQK * sizeof(float), stream>>>(
            scores, tau, 4, sec, NQK, tvs[sec], tis[sec], accs[sec]);
        k_sense_emit<KQK><<<T, 256, 0, stream>>>(
            xn1, tvs[sec], tis[sec], embs[sec], ws_[sec], dsts[sec]);
    }

    // ---- 4. attention (per b,h: QK^T -> causal softmax -> P V) ----
    k_pack_qkv<<<(T * D + 255) / 256, 256, 0, stream>>>(Qf, Kf, Vf, qbf, kbf, vtbf);
    for (int b = 0; b < BB; ++b) {
        for (int h = 0; h < H; ++h) {
            size_t bh = (size_t)(b * H + h);
            gemm(qbf + bh * S * DH, DH, kbf + bh * S * DH, DH,
                 scores, S, S, S, DH, nullptr, 0.125f);
            k_softmax_causal<<<S, 256, 0, stream>>>(scores, pbf);
            gemm(pbf, S, vtbf + bh * DH * S, S,
                 acat + (size_t)b * S * D + h * DH, D, S, DH, S, nullptr, 1.0f);
        }
    }
    k_cvt_bf<<<(T * D + 255) / 256, 256, 0, stream>>>(acat, acatbf, T * D);
    gemm(acatbf, D, wexpT, D, aout, D, T, D, D, nullptr, 1.0f);
    k_add<<<(T * D + 255) / 256, 256, 0, stream>>>(x, aout, x1, T * D);

    // ---- 5. knowledge path ----
    k_layernorm<<<T, 256, 0, stream>>>(x1, ln2_s, ln2_b, xn2, xn2bf);
    gemm(xn2bf, D, wknowT, D, hkn, DSP, T, DSP, D, proj_know_b, 1.0f);
    k_cvt_bf<<<(T * DSP + 255) / 256, 256, 0, stream>>>(hkn, hknbf, T * DSP);
    k_tau<<<T, 256, 0, stream>>>(xn2, tau_know_k, tau_know_b, 1, tauk, 1);
    gemm(hknbf, DSP, embbf + (size_t)(NQK + NV) * DSP, DSP, scores, NKN,
         T, NKN, DSP, nullptr, 1.0f);
    k_gate_topk<KKN><<<T, 256, NKN * sizeof(float), stream>>>(
        scores, tauk, 1, 0, NKN, topvN, topiN, accN);
    k_sense_emit<KKN><<<T, 256, 0, stream>>>(xn2, topvN, topiN, know_emb, know_w, kout);
    k_add<<<(T * D + 255) / 256, 256, 0, stream>>>(x1, kout, out, T * D);

    // ---- 6. aux loss ----
    k_aux<<<(3 * NQK + NKN + 255) / 256, 256, 0, stream>>>(
        accQ, accK, accV, accN, out + (size_t)T * D);
}
